// Pointnet2MSG_32323923870363
// MI455X (gfx1250) — compile-verified, matrix operand order checked by reading, not run on hardware
//
#include <hip/hip_runtime.h>
#include <hip/hip_bf16.h>

// ---------------------------------------------------------------------------
// Types for WMMA
// ---------------------------------------------------------------------------
typedef __attribute__((ext_vector_type(16))) _Float16 v16h;
typedef __attribute__((ext_vector_type(8)))  float    v8f;

#define BN_INV 0.99999500003749981f   // 1/sqrt(1+1e-5)

static inline int ceildiv_i(long a, long b) { return (int)((a + b - 1) / b); }

// ---------------------------------------------------------------------------
// Fused GEMM + BN + ReLU using v_wmma_f32_16x16x32_f16, register-blocked:
// each wave owns a 16(M) x 64(N) strip = 4 accumulators; the A fragment is
// loaded once per 32-K step and reused by 4 back-to-back WMMAs.
// The hot loop is fully branchless: N-edge handling is done by clamping the
// W row pointer (always-valid loads) and simply not storing invalid columns.
// WMMA column n only depends on B fragments in lanes n / n+16, so invalid
// lanes never pollute stored outputs.  EXEC stays all-ones at every WMMA.
// out[M,N] = act( (A[M,K] @ W[N,K]^T) * scale + bias )
//   flags&1 : scale = BN_INV * gamma[col]   (else 1.0)
//   flags&2 : ReLU
// ---------------------------------------------------------------------------
#define GEMM_NB 4
__global__ void __launch_bounds__(256) gemm_wmma_kernel(
    const float* __restrict__ A, const float* __restrict__ W,
    const float* __restrict__ gamma, const float* __restrict__ beta,
    float* __restrict__ out, int M, int N, int K, int flags)
{
    const int lane = threadIdx.x & 31;
    const long wave = (long)blockIdx.x * (blockDim.x >> 5) + (threadIdx.x >> 5);
    const int ntN  = (N + 15) >> 4;
    const int ntM  = (M + 15) >> 4;
    const int ntNB = (ntN + GEMM_NB - 1) / GEMM_NB;     // groups of 4 N-tiles
    if (wave >= (long)ntM * ntNB) return;               // wave-uniform exit
    const int mt  = (int)(wave / ntNB);
    const int nt0 = (int)(wave % ntNB) * GEMM_NB;       // first N-tile of strip

    int rowA = mt * 16 + (lane & 15);
    if (rowA >= M) rowA = M - 1;                        // clamp (M 16-aligned in practice)
    const int g8 = (lane >> 4) * 8;                     // K-half select (WMMA A/B layout)
    const float* Arow = A + (size_t)rowA * K;

    // per-subtile column / validity / clamped W row pointer (loads always valid)
    int  colB[GEMM_NB];
    bool bok[GEMM_NB];
    const float* Brow[GEMM_NB];
#pragma unroll
    for (int u = 0; u < GEMM_NB; ++u) {
        colB[u] = (nt0 + u) * 16 + (lane & 15);
        bok[u]  = colB[u] < N;                          // covers nt0+u >= ntN too
        Brow[u] = W + (size_t)(bok[u] ? colB[u] : 0) * K;
    }

    v8f acc[GEMM_NB];
#pragma unroll
    for (int u = 0; u < GEMM_NB; ++u) acc[u] = (v8f){};

    int k0 = 0;
    for (; k0 + 32 <= K; k0 += 32) {                    // branchless hot loop
        __builtin_prefetch(Arow + k0 + 64, 0, 1);
        v16h a;
#pragma unroll
        for (int j = 0; j < 8; ++j) {
            a[j]     = (_Float16)Arow[k0 + g8 + j];
            a[j + 8] = (_Float16)Arow[k0 + 16 + g8 + j];
        }
#pragma unroll
        for (int u = 0; u < GEMM_NB; ++u) {
            v16h b;
#pragma unroll
            for (int j = 0; j < 8; ++j) {
                b[j]     = (_Float16)Brow[u][k0 + g8 + j];
                b[j + 8] = (_Float16)Brow[u][k0 + 16 + g8 + j];
            }
            acc[u] = __builtin_amdgcn_wmma_f32_16x16x32_f16(
                false, a, false, b, (short)0, acc[u], false, false);
        }
    }
    if (k0 < K) {   // ragged K tail (K = 9, 99, 259, 262, 515, ...): clamp+select
        v16h a;
#pragma unroll
        for (int j = 0; j < 8; ++j) {
            const int ka = k0 + g8 + j;
            const int kb = k0 + 16 + g8 + j;
            const float av0 = Arow[ka < K ? ka : 0];
            const float av1 = Arow[kb < K ? kb : 0];
            a[j]     = (_Float16)(ka < K ? av0 : 0.f);
            a[j + 8] = (_Float16)(kb < K ? av1 : 0.f);
        }
#pragma unroll
        for (int u = 0; u < GEMM_NB; ++u) {
            v16h b;
#pragma unroll
            for (int j = 0; j < 8; ++j) {
                const int ka = k0 + g8 + j;
                const int kb = k0 + 16 + g8 + j;
                const float bv0 = Brow[u][ka < K ? ka : 0];
                const float bv1 = Brow[u][kb < K ? kb : 0];
                b[j]     = (_Float16)(ka < K ? bv0 : 0.f);
                b[j + 8] = (_Float16)(kb < K ? bv1 : 0.f);
            }
            acc[u] = __builtin_amdgcn_wmma_f32_16x16x32_f16(
                false, a, false, b, (short)0, acc[u], false, false);
        }
    }

    // epilogue: scale/bias/relu + store (C/D layout: lanes16-31 -> M+8)
    const int rbase = mt * 16 + ((lane >> 4) ? 8 : 0);
#pragma unroll
    for (int u = 0; u < GEMM_NB; ++u) {
        const int cc = bok[u] ? colB[u] : 0;            // clamped param index
        float scale = 1.f;
        if (flags & 1) scale = BN_INV * gamma[cc];
        const float bias = beta ? beta[cc] : 0.f;
#pragma unroll
        for (int r = 0; r < 8; ++r) {
            const int row = rbase + r;
            if (bok[u] && row < M) {
                float v = acc[u][r] * scale + bias;
                if (flags & 2) v = fmaxf(v, 0.f);
                out[(size_t)row * N + colB[u]] = v;
            }
        }
    }
}

// ---------------------------------------------------------------------------
// Farthest point sampling: single persistent workgroup, point cloud and
// running min-distances resident in LDS (needs CDNA5's 320KB WGP LDS).
// idx[0]=0, then npoint-1 sequential argmax steps.
// ---------------------------------------------------------------------------
__global__ void __launch_bounds__(1024) fps_kernel(
    const float* __restrict__ xyz, int n, int npoint, int* __restrict__ outidx)
{
    extern __shared__ float smem[];
    float* sx = smem;              // n*3
    float* sd = sx + (size_t)n * 3;// n
    float* rv = sd + n;            // 1024
    int*   ri = (int*)(rv + 1024); // 1024
    __shared__ int s_last;

    const int tid = threadIdx.x;
    for (int p = tid; p < n; p += blockDim.x) {
        sx[p * 3 + 0] = xyz[p * 3 + 0];
        sx[p * 3 + 1] = xyz[p * 3 + 1];
        sx[p * 3 + 2] = xyz[p * 3 + 2];
        sd[p] = 1e10f;
    }
    if (tid == 0) { s_last = 0; outidx[0] = 0; }
    __syncthreads();

    for (int i = 1; i < npoint; ++i) {
        const int last = s_last;
        const float lx = sx[last * 3], ly = sx[last * 3 + 1], lz = sx[last * 3 + 2];
        float bv = -1.f; int bi = n - 1;
        for (int p = tid; p < n; p += blockDim.x) {
            const float dx = sx[p * 3] - lx, dy = sx[p * 3 + 1] - ly, dz = sx[p * 3 + 2] - lz;
            float m = sd[p];
            const float d = dx * dx + dy * dy + dz * dz;
            m = d < m ? d : m;
            sd[p] = m;
            if (m > bv) { bv = m; bi = p; }
        }
        rv[tid] = bv; ri[tid] = bi;
        __syncthreads();
        for (int s = blockDim.x >> 1; s > 0; s >>= 1) {
            if (tid < s) {
                if (rv[tid + s] > rv[tid] ||
                    (rv[tid + s] == rv[tid] && ri[tid + s] < ri[tid])) {
                    rv[tid] = rv[tid + s]; ri[tid] = ri[tid + s];
                }
            }
            __syncthreads();
        }
        if (tid == 0) { s_last = ri[0]; outidx[i] = ri[0]; }
        __syncthreads();
    }
}

// ---------------------------------------------------------------------------
// Ball query: first nsample indices (ascending) within radius; pad with first
// hit (or 0).  One thread per center, xyz tiled through LDS.
// ---------------------------------------------------------------------------
#define BQ_TILE 512
__global__ void __launch_bounds__(256) ball_query_kernel(
    const float* __restrict__ xyz, const float* __restrict__ new_xyz,
    int n, int m, float r2, int nsample, int* __restrict__ out)
{
    __shared__ float sx[BQ_TILE * 3];
    const int c = blockIdx.x * blockDim.x + threadIdx.x;
    float cx = 0.f, cy = 0.f, cz = 0.f;
    if (c < m) { cx = new_xyz[c * 3]; cy = new_xyz[c * 3 + 1]; cz = new_xyz[c * 3 + 2]; }
    int cnt = 0;
    int* row = out + (size_t)(c < m ? c : 0) * nsample;
    for (int t0 = 0; t0 < n; t0 += BQ_TILE) {
        const int tl = (n - t0 < BQ_TILE) ? (n - t0) : BQ_TILE;
        __syncthreads();
        for (int j = threadIdx.x; j < tl * 3; j += blockDim.x) sx[j] = xyz[(size_t)t0 * 3 + j];
        __syncthreads();
        if (c < m && cnt < nsample) {
            for (int j = 0; j < tl && cnt < nsample; ++j) {
                const float dx = sx[j * 3] - cx, dy = sx[j * 3 + 1] - cy, dz = sx[j * 3 + 2] - cz;
                if (dx * dx + dy * dy + dz * dz < r2) row[cnt++] = t0 + j;
            }
        }
    }
    if (c < m) {
        const int first = (cnt == 0) ? 0 : row[0];
        for (int j = cnt; j < nsample; ++j) row[j] = first;
    }
}

// ---------------------------------------------------------------------------
// three_nn: 3 smallest squared distances + indices, known tiled through LDS.
// ---------------------------------------------------------------------------
__global__ void __launch_bounds__(256) three_nn_kernel(
    const float* __restrict__ unk, const float* __restrict__ kn,
    int nU, int nK, float* __restrict__ dist, int* __restrict__ idx)
{
    __shared__ float sx[BQ_TILE * 3];
    const int u = blockIdx.x * blockDim.x + threadIdx.x;
    float ux = 0.f, uy = 0.f, uz = 0.f;
    if (u < nU) { ux = unk[u * 3]; uy = unk[u * 3 + 1]; uz = unk[u * 3 + 2]; }
    float d0 = 1e30f, d1 = 1e30f, d2 = 1e30f;
    int i0 = 0, i1 = 0, i2 = 0;
    for (int t0 = 0; t0 < nK; t0 += BQ_TILE) {
        const int tl = (nK - t0 < BQ_TILE) ? (nK - t0) : BQ_TILE;
        __syncthreads();
        for (int j = threadIdx.x; j < tl * 3; j += blockDim.x) sx[j] = kn[(size_t)t0 * 3 + j];
        __syncthreads();
        if (u < nU) {
            for (int j = 0; j < tl; ++j) {
                const float dx = sx[j * 3] - ux, dy = sx[j * 3 + 1] - uy, dz = sx[j * 3 + 2] - uz;
                const float d = dx * dx + dy * dy + dz * dz;
                const int id = t0 + j;
                if (d < d0)      { d2 = d1; i2 = i1; d1 = d0; i1 = i0; d0 = d; i0 = id; }
                else if (d < d1) { d2 = d1; i2 = i1; d1 = d;  i1 = id; }
                else if (d < d2) { d2 = d;  i2 = id; }
            }
        }
    }
    if (u < nU) {
        dist[u * 3] = d0; dist[u * 3 + 1] = d1; dist[u * 3 + 2] = d2;
        idx[u * 3] = i0;  idx[u * 3 + 1] = i1;  idx[u * 3 + 2] = i2;
    }
}

// ---------------------------------------------------------------------------
// Elementwise helpers (grid-stride)
// ---------------------------------------------------------------------------
__global__ void group_kernel(const float* __restrict__ xyz, const float* __restrict__ feats,
                             const float* __restrict__ nxyz, const int* __restrict__ idx,
                             float* __restrict__ out, int P, int S, int C)
{
    const int Cin = 3 + C;
    const size_t tot = (size_t)P * S * Cin;
    for (size_t e = (size_t)blockIdx.x * blockDim.x + threadIdx.x; e < tot;
         e += (size_t)gridDim.x * blockDim.x) {
        const size_t row = e / Cin;
        const int c = (int)(e % Cin);
        const int p = (int)(row / S);
        const int id = idx[row];
        float v;
        if (c < 3) v = xyz[(size_t)id * 3 + c] - nxyz[(size_t)p * 3 + c];
        else       v = feats[(size_t)id * C + (c - 3)];
        out[e] = v;
    }
}

__global__ void maxpool_kernel(const float* __restrict__ in, float* __restrict__ out,
                               int P, int S, int C, int coff, int Ctot)
{
    const size_t tot = (size_t)P * C;
    for (size_t e = (size_t)blockIdx.x * blockDim.x + threadIdx.x; e < tot;
         e += (size_t)gridDim.x * blockDim.x) {
        const int p = (int)(e / C);
        const int c = (int)(e % C);
        float m = -1e30f;
        for (int s = 0; s < S; ++s) {
            const float v = in[((size_t)p * S + s) * C + c];
            m = v > m ? v : m;
        }
        out[(size_t)p * Ctot + coff + c] = m;
    }
}

__global__ void gather_kernel(const float* __restrict__ src, const int* __restrict__ idx,
                              float* __restrict__ dst, int M, int C)
{
    const size_t tot = (size_t)M * C;
    for (size_t e = (size_t)blockIdx.x * blockDim.x + threadIdx.x; e < tot;
         e += (size_t)gridDim.x * blockDim.x) {
        const int r = (int)(e / C), c = (int)(e % C);
        dst[e] = src[(size_t)idx[r] * C + c];
    }
}

__global__ void transpose_cn_kernel(const float* __restrict__ in, float* __restrict__ out,
                                    int N, int C)
{
    const size_t tot = (size_t)N * C;
    for (size_t e = (size_t)blockIdx.x * blockDim.x + threadIdx.x; e < tot;
         e += (size_t)gridDim.x * blockDim.x) {
        const int n = (int)(e / C), c = (int)(e % C);
        out[e] = in[(size_t)c * N + n];
    }
}

__global__ void concat_cols_kernel(const float* __restrict__ A, int Ca,
                                   const float* __restrict__ B, int Cb,
                                   float* __restrict__ out, int M)
{
    const int C = Ca + Cb;
    const size_t tot = (size_t)M * C;
    for (size_t e = (size_t)blockIdx.x * blockDim.x + threadIdx.x; e < tot;
         e += (size_t)gridDim.x * blockDim.x) {
        const int r = (int)(e / C), c = (int)(e % C);
        out[e] = (c < Ca) ? A[(size_t)r * Ca + c] : B[(size_t)r * Cb + (c - Ca)];
    }
}

__global__ void interp_concat_kernel(const float* __restrict__ dist, const int* __restrict__ idx,
                                     const float* __restrict__ knf, const float* __restrict__ unkf,
                                     float* __restrict__ out, int M, int CK, int CU)
{
    const int C = CK + CU;
    const size_t tot = (size_t)M * C;
    for (size_t e = (size_t)blockIdx.x * blockDim.x + threadIdx.x; e < tot;
         e += (size_t)gridDim.x * blockDim.x) {
        const int r = (int)(e / C), c = (int)(e % C);
        float v;
        if (c < CK) {
            const float w0 = 1.f / (dist[r * 3]     + 1e-8f);
            const float w1 = 1.f / (dist[r * 3 + 1] + 1e-8f);
            const float w2 = 1.f / (dist[r * 3 + 2] + 1e-8f);
            const float ws = w0 + w1 + w2;
            v = (w0 * knf[(size_t)idx[r * 3]     * CK + c] +
                 w1 * knf[(size_t)idx[r * 3 + 1] * CK + c] +
                 w2 * knf[(size_t)idx[r * 3 + 2] * CK + c]) / ws;
        } else {
            v = unkf[(size_t)r * CU + (c - CK)];
        }
        out[e] = v;
    }
}

// Attention fuse: prim <- a0*pre + a1*prim, a = softmax(Wa @ [pre;prim] + ba)
__global__ void att_fuse_kernel(const float* __restrict__ pre, float* __restrict__ prim,
                                const float* __restrict__ Wa, const float* __restrict__ ba,
                                int M, int C)
{
    const int r = blockIdx.x * blockDim.x + threadIdx.x;
    if (r >= M) return;
    float l0 = ba[0], l1 = ba[1];
    for (int c = 0; c < C; ++c) {
        const float p = pre[(size_t)r * C + c];
        const float q = prim[(size_t)r * C + c];
        l0 += Wa[c] * p + Wa[C + c] * q;
        l1 += Wa[2 * C + c] * p + Wa[3 * C + c] * q;
    }
    const float mx = fmaxf(l0, l1);
    const float e0 = __expf(l0 - mx), e1 = __expf(l1 - mx);
    const float a0 = e0 / (e0 + e1), a1 = e1 / (e0 + e1);
    for (int c = 0; c < C; ++c) {
        const size_t o = (size_t)r * C + c;
        prim[o] = a0 * pre[o] + a1 * prim[o];
    }
}

// logits [N,C] -> out[c*N+n] softmax over C
__global__ void cls_softmax_kernel(const float* __restrict__ logits, float* __restrict__ out,
                                   int N, int C)
{
    const int n = blockIdx.x * blockDim.x + threadIdx.x;
    if (n >= N) return;
    float tmp[32];
    float mx = -1e30f;
    for (int c = 0; c < C; ++c) { tmp[c] = logits[(size_t)n * C + c]; mx = fmaxf(mx, tmp[c]); }
    float s = 0.f;
    for (int c = 0; c < C; ++c) { tmp[c] = __expf(tmp[c] - mx); s += tmp[c]; }
    const float inv = 1.f / s;
    for (int c = 0; c < C; ++c) out[(size_t)c * N + n] = tmp[c] * inv;
}

// ---------------------------------------------------------------------------
// Host orchestration
// ---------------------------------------------------------------------------
struct MLPLayer { const float* W; const float* g; const float* b; int cin; int cout; };

extern "C" void kernel_launch(void* const* d_in, const int* in_sizes, int n_in,
                              void* d_out, int out_size, void* d_ws, size_t ws_size,
                              hipStream_t stream)
{
    (void)in_sizes; (void)n_in; (void)out_size; (void)ws_size;
    const int N = 16384;

    // ---- parse inputs (setup_inputs insertion order, depth-first) ----
    int ip = 0;
    const float* xyz_seq  = (const float*)d_in[ip++];   // [1,2,N,3]
    const float* feat_seq = (const float*)d_in[ip++];   // [1,2,6,N]
    auto nextp = [&]() { return (const float*)d_in[ip++]; };
    auto take_mlp = [&](MLPLayer* L, const int* dims, int nl) {
        for (int l = 0; l < nl; ++l) {
            L[l].W = nextp(); L[l].g = nextp(); L[l].b = nextp();
            L[l].cin = dims[l]; L[l].cout = dims[l + 1];
        }
    };
    static const int sa_dims[4][2][4] = {
        {{9, 16, 16, 32},   {9, 32, 32, 64}},
        {{99, 64, 64, 128}, {99, 64, 96, 128}},
        {{259, 128, 196, 256}, {259, 128, 196, 256}},
        {{515, 256, 256, 512}, {515, 256, 384, 512}}};
    MLPLayer saL[4][2][3];
    for (int m = 0; m < 4; ++m)
        for (int s = 0; s < 2; ++s) take_mlp(saL[m][s], sa_dims[m][s], 3);
    static const int d_tem1[3] = {259, 256, 768};  MLPLayer tem1[2]; take_mlp(tem1, d_tem1, 2);
    static const int d_tem2[3] = {515, 512, 1536}; MLPLayer tem2[2]; take_mlp(tem2, d_tem2, 2);
    const float* f1aW = nextp(); const float* f1ab = nextp();
    const float* f1W  = nextp(); const float* f1b  = nextp();
    const float* f2aW = nextp(); const float* f2ab = nextp();
    const float* f2W  = nextp(); const float* f2b  = nextp();
    static const int d_tfp1[2] = {1280, 256}; MLPLayer tfp1[1]; take_mlp(tfp1, d_tfp1, 1);
    static const int d_tfp2[2] = {1536, 512}; MLPLayer tfp2[1]; take_mlp(tfp2, d_tfp2, 1);
    static const int fp_dims[4][3] = {
        {262, 128, 128}, {608, 256, 256}, {768, 512, 512}, {1536, 512, 512}};
    MLPLayer fpL[4][2];
    for (int i = 0; i < 4; ++i) take_mlp(fpL[i], fp_dims[i], 2);
    static const int d_cls[2] = {128, 128}; MLPLayer clshL[1]; take_mlp(clshL, d_cls, 1);
    const float* clsW = nextp(); const float* clsb = nextp();

    // ---- workspace bump allocator ----
    char* wbase = (char*)d_ws;
    size_t woff = 0;
    auto alloc  = [&](size_t bytes) -> void* {
        bytes = (bytes + 255) & ~(size_t)255;
        void* r = wbase + woff; woff += bytes; return r;
    };
    auto allocf = [&](size_t n) -> float* { return (float*)alloc(n * 4); };
    auto alloci = [&](size_t n) -> int*   { return (int*)alloc(n * 4); };

    // persistent temporal state
    float* pre1 = allocf((size_t)1024 * 768);
    float* pre2 = allocf((size_t)256 * 1536);
    float* cx1  = allocf((size_t)1024 * 3);
    float* cx2  = allocf((size_t)256 * 3);
    float* s1f  = allocf((size_t)1024 * 256);
    float* s2f  = allocf((size_t)256 * 512);
    // big reusable buffers
    const size_t BUF_ELEMS  = (size_t)34 * 1024 * 1024;  // 136 MB: fits 524288x64
    const size_t GBUF_ELEMS = (size_t)14 * 1024 * 1024;  //  56 MB: fits 131072x99
    float* bufA = allocf(BUF_ELEMS);
    float* bufB = allocf(BUF_ELEMS);
    float* gbuf = allocf(GBUF_ELEMS);
    const size_t mark = woff;

    auto grid1 = [](size_t tot) -> int {
        size_t b = (tot + 255) / 256; return (int)(b > 8192 ? 8192 : (b ? b : 1));
    };
    auto gemm = [&](const float* A, const float* W, const float* g, const float* b,
                    float* out, int M, int Nn, int K, bool bn, bool relu) {
        const long ntM  = (M + 15) / 16;
        const long ntNB = ((Nn + 15) / 16 + GEMM_NB - 1) / GEMM_NB;
        const long tiles = ntM * ntNB;
        gemm_wmma_kernel<<<ceildiv_i(tiles, 8), 256, 0, stream>>>(
            A, W, g, b, out, M, Nn, K, (bn ? 1 : 0) | (relu ? 2 : 0));
    };
    auto run_mlp = [&](const MLPLayer* L, int nl, const float* X, int M,
                       float* final_out) -> const float* {
        const float* cur = X;
        for (int l = 0; l < nl; ++l) {
            float* dst = (l == nl - 1 && final_out) ? final_out : ((l & 1) ? bufB : bufA);
            gemm(cur, L[l].W, L[l].g, L[l].b, dst, M, L[l].cout, L[l].cin, true, true);
            cur = dst;
        }
        return cur;
    };
    auto sa_msg = [&](const float* xyzi, int n, const float* fi, int Cf, int P,
                      float r0, float r1, int S0, int S1, MLPLayer L[2][3],
                      float* nxyz, float* nfeat) {
        const int Ctot = L[0][2].cout + L[1][2].cout;
        int* fidx = alloci(P);
        fps_kernel<<<1, 1024, (size_t)n * 16 + 8192, stream>>>(xyzi, n, P, fidx);
        gather_kernel<<<grid1((size_t)P * 3), 256, 0, stream>>>(xyzi, fidx, nxyz, P, 3);
        const float rr[2] = {r0, r1}; const int SS[2] = {S0, S1};
        int coff = 0;
        for (int s = 0; s < 2; ++s) {
            const int S = SS[s], M = P * S, Cin = 3 + Cf;
            int* idx = alloci((size_t)M);
            ball_query_kernel<<<ceildiv_i(P, 256), 256, 0, stream>>>(
                xyzi, nxyz, n, P, rr[s] * rr[s], S, idx);
            group_kernel<<<grid1((size_t)M * Cin), 256, 0, stream>>>(
                xyzi, fi, nxyz, idx, gbuf, P, S, Cf);
            const float* mo = run_mlp(L[s], 3, gbuf, M, nullptr);
            maxpool_kernel<<<grid1((size_t)P * L[s][2].cout), 256, 0, stream>>>(
                mo, nfeat, P, S, L[s][2].cout, coff, Ctot);
            coff += L[s][2].cout;
        }
    };
    auto sa_single = [&](const MLPLayer* L, int nl, const float* xyzi, int n,
                         const float* fi, int Cf, int P, float radius, int S,
                         const float* given_nx, float* nx_out, float* sampled_out,
                         float* pooled) {
        const float* nx;
        if (!given_nx) {
            int* fidx = alloci(P);
            fps_kernel<<<1, 1024, (size_t)n * 16 + 8192, stream>>>(xyzi, n, P, fidx);
            gather_kernel<<<grid1((size_t)P * 3), 256, 0, stream>>>(xyzi, fidx, nx_out, P, 3);
            gather_kernel<<<grid1((size_t)P * Cf), 256, 0, stream>>>(fi, fidx, sampled_out, P, Cf);
            nx = nx_out;
        } else nx = given_nx;
        const int M = P * S, Cin = 3 + Cf;
        int* idx = alloci((size_t)M);
        ball_query_kernel<<<ceildiv_i(P, 256), 256, 0, stream>>>(
            xyzi, nx, n, P, radius * radius, S, idx);
        group_kernel<<<grid1((size_t)M * Cin), 256, 0, stream>>>(xyzi, fi, nx, idx, gbuf, P, S, Cf);
        const float* mo = run_mlp(L, nl, gbuf, M, nullptr);
        maxpool_kernel<<<grid1((size_t)P * L[nl - 1].cout), 256, 0, stream>>>(
            mo, pooled, P, S, L[nl - 1].cout, 0, L[nl - 1].cout);
    };
    auto fp_mod = [&](const MLPLayer* L, int nl, const float* unk, int nU,
                      const float* kn, int nK, const float* unkf, int CU,
                      const float* knf, int CK, float* out) {
        float* dist = allocf((size_t)nU * 3);
        int* idx = alloci((size_t)nU * 3);
        three_nn_kernel<<<ceildiv_i(nU, 256), 256, 0, stream>>>(unk, kn, nU, nK, dist, idx);
        interp_concat_kernel<<<grid1((size_t)nU * (CK + CU)), 256, 0, stream>>>(
            dist, idx, knf, unkf, gbuf, nU, CK, CU);
        run_mlp(L, nl, gbuf, nU, out);
    };

    float* outp = (float*)d_out;    // [1,2,20,N]
    for (int t = 0; t < 2; ++t) {
        woff = mark;
        const float* xyz0 = xyz_seq + (size_t)t * N * 3;
        float* feats0 = allocf((size_t)N * 6);
        transpose_cn_kernel<<<grid1((size_t)N * 6), 256, 0, stream>>>(
            feat_seq + (size_t)t * 6 * N, feats0, N, 6);

        // SA-MSG levels 1 and 2
        float* lx1 = allocf((size_t)16384 * 3); float* lf1 = allocf((size_t)16384 * 96);
        sa_msg(xyz0, N, feats0, 6, 16384, 0.5f, 1.0f, 16, 32, saL[0], lx1, lf1);
        float* lx2 = allocf((size_t)4096 * 3);  float* lf2 = allocf((size_t)4096 * 256);
        sa_msg(lx1, 16384, lf1, 96, 4096, 1.0f, 2.0f, 16, 32, saL[1], lx2, lf2);

        // temporal stage
        float* prim1  = allocf((size_t)1024 * 768);
        float* featl1 = allocf((size_t)1024 * 512);
        float* prim2  = allocf((size_t)256 * 1536);
        float* featl2 = allocf((size_t)256 * 1024);
        if (t == 0) {
            sa_single(tem1, 2, lx2, 4096, lf2, 256, 1024, 3.0f, 32, nullptr, cx1, s1f, prim1);
            gemm(prim1, f1W, nullptr, f1b, featl1, 1024, 512, 768, false, false);
            sa_single(tem2, 2, cx1, 1024, featl1, 512, 256, 6.0f, 32, nullptr, cx2, s2f, prim2);
            gemm(prim2, f2W, nullptr, f2b, featl2, 256, 1024, 1536, false, false);
            hipMemcpyAsync(pre1, prim1, (size_t)1024 * 768 * 4, hipMemcpyDeviceToDevice, stream);
            hipMemcpyAsync(pre2, prim2, (size_t)256 * 1536 * 4, hipMemcpyDeviceToDevice, stream);
        } else {
            float* exx = allocf((size_t)5120 * 3);
            float* exf = allocf((size_t)5120 * 256);
            hipMemcpyAsync(exx, lx2, (size_t)4096 * 3 * 4, hipMemcpyDeviceToDevice, stream);
            hipMemcpyAsync(exx + (size_t)4096 * 3, cx1, (size_t)1024 * 3 * 4, hipMemcpyDeviceToDevice, stream);
            hipMemcpyAsync(exf, lf2, (size_t)4096 * 256 * 4, hipMemcpyDeviceToDevice, stream);
            hipMemcpyAsync(exf + (size_t)4096 * 256, s1f, (size_t)1024 * 256 * 4, hipMemcpyDeviceToDevice, stream);
            sa_single(tem1, 2, exx, 5120, exf, 256, 1024, 3.0f, 32, cx1, nullptr, nullptr, prim1);
            att_fuse_kernel<<<ceildiv_i(1024, 128), 128, 0, stream>>>(pre1, prim1, f1aW, f1ab, 1024, 768);
            hipMemcpyAsync(pre1, prim1, (size_t)1024 * 768 * 4, hipMemcpyDeviceToDevice, stream);
            gemm(prim1, f1W, nullptr, f1b, featl1, 1024, 512, 768, false, false);
            float* exx1 = allocf((size_t)1280 * 3);
            float* exf1 = allocf((size_t)1280 * 512);
            hipMemcpyAsync(exx1, cx1, (size_t)1024 * 3 * 4, hipMemcpyDeviceToDevice, stream);
            hipMemcpyAsync(exx1 + (size_t)1024 * 3, cx2, (size_t)256 * 3 * 4, hipMemcpyDeviceToDevice, stream);
            hipMemcpyAsync(exf1, featl1, (size_t)1024 * 512 * 4, hipMemcpyDeviceToDevice, stream);
            hipMemcpyAsync(exf1 + (size_t)1024 * 512, s2f, (size_t)256 * 512 * 4, hipMemcpyDeviceToDevice, stream);
            sa_single(tem2, 2, exx1, 1280, exf1, 512, 256, 6.0f, 32, cx2, nullptr, nullptr, prim2);
            att_fuse_kernel<<<ceildiv_i(256, 128), 128, 0, stream>>>(pre2, prim2, f2aW, f2ab, 256, 1536);
            hipMemcpyAsync(pre2, prim2, (size_t)256 * 1536 * 4, hipMemcpyDeviceToDevice, stream);
            gemm(prim2, f2W, nullptr, f2b, featl2, 256, 1024, 1536, false, false);
        }

        // temporal feature propagation
        float* curx = allocf((size_t)1024 * 512);
        fp_mod(tfp2, 1, cx1, 1024, cx2, 256, featl1, 512, featl2, 1024, curx);
        float* ffp1 = allocf((size_t)1024 * 1024);
        concat_cols_kernel<<<grid1((size_t)1024 * 1024), 256, 0, stream>>>(
            curx, 512, featl1, 512, ffp1, 1024);
        float* lf2b = allocf((size_t)4096 * 256);
        fp_mod(tfp1, 1, lx2, 4096, cx1, 1024, lf2, 256, ffp1, 1024, lf2b);

        // SA-MSG levels 3 and 4
        float* lx3 = allocf((size_t)1024 * 3); float* lf3 = allocf((size_t)1024 * 512);
        sa_msg(lx2, 4096, lf2b, 256, 1024, 2.0f, 4.0f, 16, 32, saL[2], lx3, lf3);
        float* lx4 = allocf((size_t)256 * 3);  float* lf4 = allocf((size_t)256 * 1024);
        sa_msg(lx3, 1024, lf3, 512, 256, 4.0f, 8.0f, 16, 32, saL[3], lx4, lf4);

        // FP decoder
        float* dlf3 = allocf((size_t)1024 * 512);
        fp_mod(fpL[3], 2, lx3, 1024, lx4, 256, lf3, 512, lf4, 1024, dlf3);
        float* dlf2 = allocf((size_t)4096 * 512);
        fp_mod(fpL[2], 2, lx2, 4096, lx3, 1024, lf2b, 256, dlf3, 512, dlf2);
        float* dlf1 = allocf((size_t)16384 * 256);
        fp_mod(fpL[1], 2, lx1, 16384, lx2, 4096, lf1, 96, dlf2, 512, dlf1);
        float* dlf0 = allocf((size_t)16384 * 128);
        fp_mod(fpL[0], 2, xyz0, 16384, lx1, 16384, feats0, 6, dlf1, 256, dlf0);

        // classifier head + softmax over classes
        float* hid = allocf((size_t)16384 * 128);
        gemm(dlf0, clshL[0].W, clshL[0].g, clshL[0].b, hid, 16384, 128, 128, true, true);
        float* logits = allocf((size_t)16384 * 20);
        gemm(hid, clsW, nullptr, clsb, logits, 16384, 20, 128, false, false);
        cls_softmax_kernel<<<ceildiv_i(N, 256), 256, 0, stream>>>(
            logits, outp + (size_t)t * 20 * N, N, 20);
    }
}
